// PointNet2ClsMsg_36112085025338
// MI455X (gfx1250) — matrix operand order repaired and checked by reading, not compiled, not run on hardware
//
#include <hip/hip_runtime.h>
#include <hip/hip_bf16.h>

// PointNet++ MSG classifier for gfx1250 (MI455X).
// Mixed precision: f16 activations/weights, f32 WMMA accumulation via
// v_wmma_f32_16x16x32_f16. Each wave computes a 16x32 output block (2 WMMAs
// sharing one A fragment -> halves A traffic). BN stats in f32 with
// hierarchical (LDS -> global) atomics. Assumes ws_size >= ~280MB.

typedef _Float16 f16;
typedef _Float16 v16h __attribute__((ext_vector_type(16)));
typedef float    v8f  __attribute__((ext_vector_type(8)));
typedef unsigned int u32x4 __attribute__((ext_vector_type(4)));

union V16F { v16h h; u32x4 q[2]; };

#define BN_EPS 1e-5f

// ---------------------------------------------------------------- kernels

// xyz (B,6,N) -> pts f32 (B,N,3), nrm f16 (B,N,3)
__global__ void k_split(const float* __restrict__ xyz, float* __restrict__ pts,
                        f16* __restrict__ nrm, int Bn, int N)
{
  long t = (long)blockIdx.x * blockDim.x + threadIdx.x;
  if (t >= (long)Bn * N) return;
  int b = (int)(t / N), n = (int)(t % N);
  const float* src = xyz + (size_t)b * 6 * N;
  float* pd = pts + ((size_t)b * N + n) * 3;
  f16*   nd = nrm + ((size_t)b * N + n) * 3;
  pd[0] = src[0 * N + n]; pd[1] = src[1 * N + n]; pd[2] = src[2 * N + n];
  nd[0] = (f16)src[3 * N + n]; nd[1] = (f16)src[4 * N + n]; nd[2] = (f16)src[5 * N + n];
}

// W f32 (Cout,Cin) -> Wh f16 (Coutp,Kpad) zero padded
__global__ void k_cvt_w(const float* __restrict__ W, f16* __restrict__ Wh,
                        int Cout, int Cin, int Coutp, int Kpad)
{
  long t = (long)blockIdx.x * blockDim.x + threadIdx.x;
  if (t >= (long)Coutp * Kpad) return;
  int r = (int)(t / Kpad), c = (int)(t % Kpad);
  float v = (r < Cout && c < Cin) ? W[(size_t)r * Cin + c] : 0.f;
  Wh[t] = (f16)v;
}

__global__ void k_zero_f(float* __restrict__ p, int n)
{
  int t = blockIdx.x * blockDim.x + threadIdx.x;
  if (t < n) p[t] = 0.f;
}

// Farthest point sampling: one workgroup per batch. Matches JAX scan semantics:
// first recorded index is 0; argmax breaks ties toward lower index.
__global__ void k_fps(const float* __restrict__ pts, int N, int S,
                      int* __restrict__ outIdx, float* __restrict__ newXyz)
{
  int b = blockIdx.x;
  const float* P = pts + (size_t)b * N * 3;
  __shared__ float dist[4096];
  __shared__ float rv[256];
  __shared__ int   ri[256];
  __shared__ float fpt[3];
  __shared__ int   sfar;
  int tid = threadIdx.x;
  for (int j = tid; j < N; j += blockDim.x) dist[j] = 1e10f;
  if (tid == 0) sfar = 0;
  __syncthreads();
  for (int i = 0; i < S; ++i) {
    if (tid == 0) {
      int far = sfar;
      outIdx[(size_t)b * S + i] = far;
      float px = P[far * 3 + 0], py = P[far * 3 + 1], pz = P[far * 3 + 2];
      fpt[0] = px; fpt[1] = py; fpt[2] = pz;
      float* nx = newXyz + ((size_t)b * S + i) * 3;
      nx[0] = px; nx[1] = py; nx[2] = pz;
    }
    __syncthreads();
    float fx = fpt[0], fy = fpt[1], fz = fpt[2];
    float bv = -1.f; int bi = 0;
    for (int j = tid; j < N; j += blockDim.x) {
      float dx = P[j * 3 + 0] - fx, dy = P[j * 3 + 1] - fy, dz = P[j * 3 + 2] - fz;
      float d = dx * dx + dy * dy + dz * dz;
      float nd = fminf(dist[j], d);
      dist[j] = nd;
      if (nd > bv) { bv = nd; bi = j; }
    }
    rv[tid] = bv; ri[tid] = bi;
    __syncthreads();
    for (int s2 = 128; s2 > 0; s2 >>= 1) {
      if (tid < s2) {
        if (rv[tid + s2] > rv[tid] ||
            (rv[tid + s2] == rv[tid] && ri[tid + s2] < ri[tid])) {
          rv[tid] = rv[tid + s2]; ri[tid] = ri[tid + s2];
        }
      }
      __syncthreads();
    }
    if (tid == 0) sfar = ri[0];
    __syncthreads();
  }
}

// Ball query: first K in-range indices in ascending order, pad with first
// (or N-1 if empty, matching JAX clamp-on-gather).
__global__ void k_ball(const float* __restrict__ pts, const float* __restrict__ nx,
                       int* __restrict__ idx, int Bn, int N, int S, int K, float r2)
{
  int t = blockIdx.x * blockDim.x + threadIdx.x;
  if (t >= Bn * S) return;
  int b = t / S;
  const float* P = pts + (size_t)b * N * 3;
  float cx = nx[(size_t)t * 3 + 0], cy = nx[(size_t)t * 3 + 1], cz = nx[(size_t)t * 3 + 2];
  int* out = idx + (size_t)t * K;
  int cnt = 0, first = -1;
  for (int j = 0; j < N && cnt < K; ++j) {
    float dx = P[j * 3 + 0] - cx, dy = P[j * 3 + 1] - cy, dz = P[j * 3 + 2] - cz;
    float d = dx * dx + dy * dy + dz * dz;
    if (d <= r2) { if (first < 0) first = j; out[cnt++] = j; }
  }
  int fill = (first < 0) ? (N - 1) : first;
  for (; cnt < K; ++cnt) out[cnt] = fill;
}

// Build grouped GEMM input: A[row, 0..C-1]=feats[idx], A[row, C..C+2]=gxyz, rest 0.
__global__ void k_group(const f16* __restrict__ feats, const float* __restrict__ pts,
                        const float* __restrict__ nx, const int* __restrict__ idx,
                        f16* __restrict__ A, int Bn, int N, int S, int K, int C, int Kpad)
{
  long t = (long)blockIdx.x * blockDim.x + threadIdx.x;
  long total = (long)Bn * S * K * Kpad;
  if (t >= total) return;
  int c = (int)(t % Kpad);
  long row = t / Kpad;
  long bs = row / K;
  int s = (int)(bs % S), b = (int)(bs / S);
  int id = idx[row];
  float v = 0.f;
  if (c < C) {
    v = (float)feats[((size_t)b * N + id) * C + c];
  } else if (c < C + 3) {
    int d = c - C;
    v = pts[((size_t)b * N + id) * 3 + d] - nx[((size_t)b * S + s) * 3 + d];
  }
  A[(size_t)row * Kpad + c] = (f16)v;
}

// sa3 grouping: concat([x2, p2]) -> (B*128, Kpad)
__global__ void k_group3(const f16* __restrict__ p2, const float* __restrict__ x2,
                         f16* __restrict__ A, int Bn, int Kpad)
{
  long t = (long)blockIdx.x * blockDim.x + threadIdx.x;
  long total = (long)Bn * 128 * Kpad;
  if (t >= total) return;
  int c = (int)(t % Kpad);
  long row = t / Kpad;  // b*128 + s
  float v = 0.f;
  if (c < 3)        v = x2[(size_t)row * 3 + c];
  else if (c < 643) v = (float)p2[(size_t)row * 640 + (c - 3)];
  A[(size_t)row * Kpad + c] = (f16)v;
}

// WMMA GEMM: Y[M,Nout] = X[M,Kpad] * Wh[Coutp,Kpad]^T + bias, f32 accumulate.
// Each wave: one 16x32 output block (one A fragment, two B fragments, two
// WMMAs per k-step). Optionally accumulates per-channel sum/sumsq (for BN)
// via LDS then global atomics.
// Fragment layouts used here (wave32, 16x16x32 f16; k = 16*g + s for both A and B):
//  - A per lane: one contiguous 32B run at half offset g*16
//  - B per lane: one contiguous 32B run at half offset g*16
// A row / B row indices are CLAMPED (not guarded) so the hot loop has zero
// exec-mask branches; out-of-range rows/cols are masked in the epilogue.
__global__ void __launch_bounds__(256)
k_gemm(const f16* __restrict__ X, const f16* __restrict__ Wh,
       f16* __restrict__ Yh, float* __restrict__ Yf,
       const float* __restrict__ bias,
       float* __restrict__ sums, float* __restrict__ sumsq,
       int M, int Kpad, int Nout, int Nld, int Coutp)
{
  __shared__ float ssum[32];
  __shared__ float ssq[32];
  if (threadIdx.x < 32) { ssum[threadIdx.x] = 0.f; ssq[threadIdx.x] = 0.f; }
  __syncthreads();

  int wave = threadIdx.x >> 5;
  int lane = threadIdx.x & 31;
  int mtile = blockIdx.x * 8 + wave;
  int g  = lane >> 4;     // lane half
  int mr = lane & 15;     // A row / B,D column within tile

  int arow = mtile * 16 + mr;
  if (arow >= M) arow = M - 1;                  // clamp -> branch-free loads
  const f16* Ap = X + (size_t)arow * Kpad + g * 16;

  int ncol0 = blockIdx.y * 32 + mr;
  int ncol1 = ncol0 + 16;
  int brow1 = (ncol1 < Coutp) ? ncol1 : (Coutp - 1);  // clamp (fc3: Coutp=48)
  const f16* Bp0 = Wh + (size_t)ncol0 * Kpad + g * 16;
  const f16* Bp1 = Wh + (size_t)brow1 * Kpad + g * 16;

  v8f acc0 = {};
  v8f acc1 = {};
#pragma unroll 2
  for (int kb = 0; kb < Kpad; kb += 32) {
    V16F a, b0, b1;
    a.q[0]  = *(const u32x4*)(Ap + kb);        // halves g*16  .. g*16+7
    a.q[1]  = *(const u32x4*)(Ap + kb + 8);    // halves g*16+8.. g*16+15
    b0.q[0] = *(const u32x4*)(Bp0 + kb);
    b0.q[1] = *(const u32x4*)(Bp0 + kb + 8);
    b1.q[0] = *(const u32x4*)(Bp1 + kb);
    b1.q[1] = *(const u32x4*)(Bp1 + kb + 8);
    acc0 = __builtin_amdgcn_wmma_f32_16x16x32_f16(
        false, a.h, false, b0.h, (short)0, acc0, false, false);
    acc1 = __builtin_amdgcn_wmma_f32_16x16x32_f16(
        false, a.h, false, b1.h, (short)0, acc1, false, false);
  }

  bool ncok0 = ncol0 < Nout;
  bool ncok1 = ncol1 < Nout;
  float bv0 = ncok0 ? bias[ncol0] : 0.f;
  float bv1 = ncok1 ? bias[ncol1] : 0.f;
  float s0 = 0.f, sq0 = 0.f, s1 = 0.f, sq1 = 0.f;
#pragma unroll
  for (int v = 0; v < 8; ++v) {
    int mrow = mtile * 16 + g * 8 + v;   // D: VGPR v -> M = v + 8*g
    if (mrow < M) {
      if (ncok0) {
        float val = acc0[v] + bv0;
        if (Yh) Yh[(size_t)mrow * Nld + ncol0] = (f16)val;
        if (Yf) Yf[(size_t)mrow * Nld + ncol0] = val;
        s0 += val; sq0 += val * val;
      }
      if (ncok1) {
        float val = acc1[v] + bv1;
        if (Yh) Yh[(size_t)mrow * Nld + ncol1] = (f16)val;
        if (Yf) Yf[(size_t)mrow * Nld + ncol1] = val;
        s1 += val; sq1 += val * val;
      }
    }
  }
  if (sums) {
    s0  += __shfl_xor(s0, 16, 32);   // combine lane pair sharing column
    sq0 += __shfl_xor(sq0, 16, 32);
    s1  += __shfl_xor(s1, 16, 32);
    sq1 += __shfl_xor(sq1, 16, 32);
    if (g == 0) {
      if (ncok0) { atomicAdd(&ssum[mr],      s0); atomicAdd(&ssq[mr],      sq0); }
      if (ncok1) { atomicAdd(&ssum[16 + mr], s1); atomicAdd(&ssq[16 + mr], sq1); }
    }
  }
  __syncthreads();
  if (sums && threadIdx.x < 32) {
    int nc = blockIdx.y * 32 + (int)threadIdx.x;
    if (nc < Nout) {
      atomicAdd(&sums[nc],  ssum[threadIdx.x]);
      atomicAdd(&sumsq[nc], ssq[threadIdx.x]);
    }
  }
}

// scale = g * rsqrt(var+eps); shift = be - mu*scale
__global__ void k_bnfin(const float* __restrict__ sums, const float* __restrict__ sumsq,
                        const float* __restrict__ gg, const float* __restrict__ be,
                        float* __restrict__ scale, float* __restrict__ shift,
                        int Nout, float invCnt)
{
  int c = blockIdx.x * blockDim.x + threadIdx.x;
  if (c >= Nout) return;
  float mu  = sums[c] * invCnt;
  float var = sumsq[c] * invCnt - mu * mu;
  float inv = rsqrtf(var + BN_EPS);
  float sc  = gg[c] * inv;
  scale[c] = sc;
  shift[c] = be[c] - mu * sc;
}

// in-place BN affine + relu on f16 tensor (M x C, ld == C)
__global__ void k_bnrelu(f16* __restrict__ Y, const float* __restrict__ scale,
                         const float* __restrict__ shift, long total, int C)
{
  long t = (long)blockIdx.x * blockDim.x + threadIdx.x;
  if (t >= total) return;
  int c = (int)(t % C);
  float v = (float)Y[t] * scale[c] + shift[c];
  Y[t] = (f16)fmaxf(v, 0.f);
}

// fused BN + relu + max over K group points; writes f16 and/or f32 destination
__global__ void k_maxpool(const f16* __restrict__ Y, const float* __restrict__ scale,
                          const float* __restrict__ shift,
                          f16* __restrict__ dsth, float* __restrict__ dstf,
                          int BS, int K, int Nout, int dstld, int coff)
{
  long t = (long)blockIdx.x * blockDim.x + threadIdx.x;
  if (t >= (long)BS * Nout) return;
  int c = (int)(t % Nout);
  long bs = t / Nout;
  const f16* p = Y + (size_t)bs * K * Nout + c;
  float sc = scale[c], sh = shift[c];
  float m = 0.f;  // relu makes everything >= 0
  for (int k = 0; k < K; ++k) {
    float v = fmaxf((float)p[(size_t)k * Nout] * sc + sh, 0.f);
    if (v > m) m = v;
  }
  if (dsth) dsth[(size_t)bs * dstld + coff + c] = (f16)m;
  if (dstf) dstf[(size_t)bs * dstld + coff + c] = m;
}

__global__ void k_lsm(const float* __restrict__ logits, float* __restrict__ out,
                      int Bn, int C)
{
  int b = threadIdx.x;
  if (b >= Bn) return;
  const float* x = logits + (size_t)b * C;
  float mx = x[0];
  for (int c = 1; c < C; ++c) mx = fmaxf(mx, x[c]);
  float se = 0.f;
  for (int c = 0; c < C; ++c) se += expf(x[c] - mx);
  float l = logf(se);
  for (int c = 0; c < C; ++c) out[(size_t)b * C + c] = x[c] - mx - l;
}

// ---------------------------------------------------------------- host

extern "C" void kernel_launch(void* const* d_in, const int* in_sizes, int n_in,
                              void* d_out, int out_size, void* d_ws, size_t ws_size,
                              hipStream_t stream)
{
  (void)in_sizes; (void)n_in; (void)out_size; (void)ws_size;
  const int B = 8, N1 = 4096, S1 = 512, S2 = 128;

  // layer tables (d_in index of W; b=+1, g=+2, be=+3), insertion-order flatten:
  // xyz, sa1[3][3], sa2[3][3], sa3[3], fc1, fc2, fc3
  static const int LD[24]  = {1,5,9, 13,17,21, 25,29,33, 37,41,45, 49,53,57,
                              61,65,69, 73,77,81, 85,89,93};
  static const int LCO[24] = {32,32,64, 64,64,128, 64,96,128, 64,64,128,
                              128,128,256, 128,128,256, 256,512,1024, 512,256,40};
  static const int LCI[24] = {6,32,32, 6,64,64, 6,64,96, 323,64,64,
                              323,128,128, 323,128,128, 643,256,512, 1024,512,256};
  int coutp[24], kpad[24]; size_t woff[24]; size_t wtot = 0;
  for (int i = 0; i < 24; ++i) {
    coutp[i] = (LCO[i] + 15) & ~15;
    kpad[i]  = (LCI[i] + 31) & ~31;
    woff[i]  = wtot;
    wtot += (size_t)coutp[i] * kpad[i];
  }

  char* base = (char*)d_ws; size_t cur = 0;
  auto alloc = [&](size_t bytes) -> char* {
    char* p = base + cur; cur += (bytes + 255) & ~(size_t)255; return p;
  };
  float* pts   = (float*)alloc((size_t)B * N1 * 3 * 4);
  f16*   nrm   = (f16*)  alloc((size_t)B * N1 * 3 * 2);
  f16*   WhB   = (f16*)  alloc(wtot * 2);
  int*   fpsI  = (int*)  alloc((size_t)B * S1 * 4);
  float* x1    = (float*)alloc((size_t)B * S1 * 3 * 4);
  float* x2    = (float*)alloc((size_t)B * S2 * 3 * 4);
  int*   ball  = (int*)  alloc((size_t)B * S1 * 128 * 4);
  f16*   p1    = (f16*)  alloc((size_t)B * S1 * 320 * 2);
  f16*   p2    = (f16*)  alloc((size_t)B * S2 * 640 * 2);
  f16*   featH = (f16*)  alloc((size_t)B * 1024 * 2);
  f16*   fcY1  = (f16*)  alloc((size_t)B * 512 * 2);
  f16*   fcY2  = (f16*)  alloc((size_t)B * 256 * 2);
  float* logit = (float*)alloc((size_t)B * 40 * 4);
  float* sums  = (float*)alloc(1024 * 4);
  float* sumsq = (float*)alloc(1024 * 4);
  float* scale = (float*)alloc(1024 * 4);
  float* shift = (float*)alloc(1024 * 4);
  f16*   buf0  = (f16*)  alloc((size_t)524288 * 128 * 2);  // 134 MB
  f16*   buf1  = (f16*)  alloc((size_t)524288 * 128 * 2);  // 134 MB

  const float* xyz = (const float*)d_in[0];
  float* outF = (float*)d_out;

  auto cdiv = [](long a, long b) -> long { return (a + b - 1) / b; };

  k_split<<<dim3((unsigned)cdiv((long)B * N1, 256)), dim3(256), 0, stream>>>(
      xyz, pts, nrm, B, N1);

  for (int i = 0; i < 24; ++i) {
    long tot = (long)coutp[i] * kpad[i];
    k_cvt_w<<<dim3((unsigned)cdiv(tot, 256)), dim3(256), 0, stream>>>(
        (const float*)d_in[LD[i]], WhB + woff[i], LCO[i], LCI[i], coutp[i], kpad[i]);
  }

  auto gemm = [&](const f16* X, int li, f16* Yh, float* Yf, long M, bool stats) {
    if (stats) {
      k_zero_f<<<dim3(4), dim3(256), 0, stream>>>(sums, 1024);
      k_zero_f<<<dim3(4), dim3(256), 0, stream>>>(sumsq, 1024);
    }
    long Mt = cdiv(M, 16);
    dim3 g((unsigned)cdiv(Mt, 8), (unsigned)cdiv(coutp[li], 32));
    k_gemm<<<g, dim3(256), 0, stream>>>(
        X, WhB + woff[li], Yh, Yf, (const float*)d_in[LD[li] + 1],
        stats ? sums : nullptr, stats ? sumsq : nullptr,
        (int)M, kpad[li], LCO[li], LCO[li], coutp[li]);
  };
  auto bnfin = [&](int li, long cnt) {
    k_bnfin<<<dim3((unsigned)cdiv(LCO[li], 256)), dim3(256), 0, stream>>>(
        sums, sumsq, (const float*)d_in[LD[li] + 2], (const float*)d_in[LD[li] + 3],
        scale, shift, LCO[li], 1.f / (float)cnt);
  };
  auto bnrelu = [&](f16* Y, long M, int C) {
    k_bnrelu<<<dim3((unsigned)cdiv(M * C, 256)), dim3(256), 0, stream>>>(
        Y, scale, shift, M * (long)C, C);
  };

  auto run_branch = [&](const float* P, const f16* F, const float* NX,
                        int N, int S, int C, int K, float r, int L0,
                        f16* pout, int pld, int coff) {
    int BS = B * S;
    k_ball<<<dim3((unsigned)cdiv(BS, 256)), dim3(256), 0, stream>>>(
        P, NX, ball, B, N, S, K, r * r);
    long M = (long)BS * K;
    long tot = M * kpad[L0];
    k_group<<<dim3((unsigned)cdiv(tot, 256)), dim3(256), 0, stream>>>(
        F, P, NX, ball, buf0, B, N, S, K, C, kpad[L0]);
    f16* X = buf0;
    for (int li = 0; li < 3; ++li) {
      f16* Y = (li & 1) ? buf0 : buf1;
      gemm(X, L0 + li, Y, nullptr, M, true);
      bnfin(L0 + li, M);
      if (li < 2) {
        bnrelu(Y, M, LCO[L0 + li]);
        X = Y;
      } else {
        k_maxpool<<<dim3((unsigned)cdiv((long)BS * LCO[L0 + li], 256)), dim3(256), 0, stream>>>(
            Y, scale, shift, pout, nullptr, BS, K, LCO[L0 + li], pld, coff);
      }
    }
  };

  // ---- SA1: npoint 512, radii {0.1,0.2,0.4}, K {16,32,128} ----
  k_fps<<<dim3(B), dim3(256), 0, stream>>>(pts, N1, S1, fpsI, x1);
  run_branch(pts, nrm, x1, N1, S1, 3,  16, 0.1f, 0, p1, 320, 0);
  run_branch(pts, nrm, x1, N1, S1, 3,  32, 0.2f, 3, p1, 320, 64);
  run_branch(pts, nrm, x1, N1, S1, 3, 128, 0.4f, 6, p1, 320, 192);

  // ---- SA2: npoint 128, radii {0.2,0.4,0.8}, K {32,64,128} ----
  k_fps<<<dim3(B), dim3(256), 0, stream>>>(x1, S1, S2, fpsI, x2);
  run_branch(x1, p1, x2, S1, S2, 320,  32, 0.2f,  9, p2, 640, 0);
  run_branch(x1, p1, x2, S1, S2, 320,  64, 0.4f, 12, p2, 640, 128);
  run_branch(x1, p1, x2, S1, S2, 320, 128, 0.8f, 15, p2, 640, 384);

  // ---- SA3: group-all over 128 points ----
  {
    long M = (long)B * 128;
    long tot = M * kpad[18];
    k_group3<<<dim3((unsigned)cdiv(tot, 256)), dim3(256), 0, stream>>>(
        p2, x2, buf0, B, kpad[18]);
    f16* X = buf0;
    for (int li = 0; li < 3; ++li) {
      f16* Y = (li & 1) ? buf0 : buf1;
      gemm(X, 18 + li, Y, nullptr, M, true);
      bnfin(18 + li, M);
      if (li < 2) {
        bnrelu(Y, M, LCO[18 + li]);
        X = Y;
      } else {
        // feat f16 for FC input + f32 to d_out (feat output, after logits block)
        k_maxpool<<<dim3((unsigned)cdiv((long)B * 1024, 256)), dim3(256), 0, stream>>>(
            Y, scale, shift, featH, outF + 320, B, 128, 1024, 1024, 0);
      }
    }
  }

  // ---- FC head ----
  gemm(featH, 21, fcY1, nullptr, 8, true); bnfin(21, 8); bnrelu(fcY1, 8, 512);
  gemm(fcY1,  22, fcY2, nullptr, 8, true); bnfin(22, 8); bnrelu(fcY2, 8, 256);
  gemm(fcY2,  23, nullptr, logit, 8, false);
  k_lsm<<<dim3(1), dim3(32), 0, stream>>>(logit, outF, 8, 40);
}
